// GraphAttnNet_55705725829963
// MI455X (gfx1250) — compile-verified, matrix operand order checked by reading, not run on hardware
//
#include <hip/hip_runtime.h>
#include <hip/hip_bf16.h>

typedef __attribute__((ext_vector_type(16))) _Float16 v16h;
typedef __attribute__((ext_vector_type(8)))  float    v8f;

#define H 128

__device__ __forceinline__ float leakyf(float x) { return x >= 0.f ? x : 0.2f * x; }
__device__ __forceinline__ float sigmoidf_(float x) { return 1.0f / (1.0f + __expf(-x)); }

// ---------------- fill ----------------
__global__ void fill_u32_kernel(unsigned* __restrict__ p, unsigned v, int n) {
    int i = blockIdx.x * blockDim.x + threadIdx.x;
    if (i < n) p[i] = v;
}

// ---------------- prep: Q[l][d] = enc_edge_W[d,:] . w_e[l], c[l], graph offsets ----------------
__global__ void prep_kernel(const float* __restrict__ encW, const float* __restrict__ encb,
                            const float* __restrict__ bew, const int* __restrict__ natoms,
                            float* __restrict__ Q, float* __restrict__ cvec,
                            int* __restrict__ goffs, int G, int L) {
    int t = threadIdx.x;
    if (t < 64 * 3) {
        int l = t >> 6, d = t & 63;
        if (l < L) {
            float s = 0.f;
            for (int k = 0; k < H; ++k) s += encW[d * H + k] * bew[l * (3 * H) + k];
            Q[l * 64 + d] = s;
        }
    }
    if (t < L) {
        float s = 0.f;
        for (int k = 0; k < H; ++k) s += encb[k] * bew[t * (3 * H) + k];
        cvec[t] = s;
    }
    if (t == 0) {
        int acc = 0;
        for (int g = 0; g < G; ++g) { goffs[g] = acc; acc += natoms[g]; }
        goffs[G] = acc;
    }
}

// ---------------- e_dot for all 3 layers: 16 lanes cooperate per edge ----------------
__global__ __launch_bounds__(256) void edot_kernel(const float* __restrict__ edge_attr,
                                                   const float* __restrict__ Q,
                                                   const float* __restrict__ cvec,
                                                   float* __restrict__ edot, int E) {
    int gid = blockIdx.x * blockDim.x + threadIdx.x;
    int e = gid >> 4;
    if (e >= E) return;
    int sub = gid & 15;
    float4 ea = *(const float4*)&edge_attr[(size_t)e * 64 + sub * 4];
    float p0, p1, p2;
    {
        float4 q0 = *(const float4*)&Q[0 * 64 + sub * 4];
        float4 q1 = *(const float4*)&Q[1 * 64 + sub * 4];
        float4 q2 = *(const float4*)&Q[2 * 64 + sub * 4];
        p0 = ea.x * q0.x + ea.y * q0.y + ea.z * q0.z + ea.w * q0.w;
        p1 = ea.x * q1.x + ea.y * q1.y + ea.z * q1.z + ea.w * q1.w;
        p2 = ea.x * q2.x + ea.y * q2.y + ea.z * q2.z + ea.w * q2.w;
    }
    #pragma unroll
    for (int m = 8; m >= 1; m >>= 1) {
        p0 += __shfl_xor(p0, m, 32);
        p1 += __shfl_xor(p1, m, 32);
        p2 += __shfl_xor(p2, m, 32);
    }
    if (sub == 0) {
        edot[(size_t)0 * E + e] = p0 + cvec[0];
        edot[(size_t)1 * E + e] = p1 + cvec[1];
        edot[(size_t)2 * E + e] = p2 + cvec[2];
    }
}

// ============ edge bucketing by src (done once, reused for all layers) ============
__global__ void count_kernel(const int* __restrict__ src, int* __restrict__ deg, int E) {
    int e = blockIdx.x * blockDim.x + threadIdx.x;
    if (e < E) atomicAdd(&deg[src[e]], 1);
}

// 3-phase exclusive scan of deg[0..n) -> nodeoff[0..n], nodeoff[n]=E
__global__ void scanA_kernel(const int* __restrict__ deg, int* __restrict__ nodeoff,
                             int* __restrict__ bsum, int n) {
    __shared__ int tmp[256];
    int i = blockIdx.x * 256 + threadIdx.x;
    int v = (i < n) ? deg[i] : 0;
    tmp[threadIdx.x] = v;
    __syncthreads();
    for (int off = 1; off < 256; off <<= 1) {
        int t = (threadIdx.x >= off) ? tmp[threadIdx.x - off] : 0;
        __syncthreads();
        tmp[threadIdx.x] += t;
        __syncthreads();
    }
    if (i < n) nodeoff[i] = tmp[threadIdx.x] - v;  // exclusive
    if (threadIdx.x == 255) bsum[blockIdx.x] = tmp[255];
}
__global__ void scanB_kernel(int* __restrict__ bsum, int nb) {
    if (threadIdx.x == 0 && blockIdx.x == 0) {
        int acc = 0;
        for (int i = 0; i < nb; ++i) { int t = bsum[i]; bsum[i] = acc; acc += t; }
    }
}
__global__ void scanC_kernel(int* __restrict__ nodeoff, const int* __restrict__ bsum,
                             int n, int E) {
    int i = blockIdx.x * 256 + threadIdx.x;
    if (i < n) nodeoff[i] += bsum[blockIdx.x];
    if (blockIdx.x == 0 && threadIdx.x == 0) nodeoff[n] = E;
}
__global__ void copy_i32_kernel(const int* __restrict__ a, int* __restrict__ b, int n) {
    int i = blockIdx.x * blockDim.x + threadIdx.x;
    if (i < n) b[i] = a[i];
}
__global__ void scatter_kernel(const int* __restrict__ src, int* __restrict__ cursor,
                               int* __restrict__ order, int E) {
    int e = blockIdx.x * blockDim.x + threadIdx.x;
    if (e >= E) return;
    int p = atomicAdd(&cursor[src[e]], 1);
    order[p] = e;
}

// ---------------- WMMA GEMM: C[M,128] = A[M,128] @ W[128,128] + bias (64-row blocks) ----------------
__global__ __launch_bounds__(256) void gemm128_wmma(const float* __restrict__ A,
                                                    const float* __restrict__ W,
                                                    const float* __restrict__ bias,
                                                    float* __restrict__ C, int M) {
    __shared__ float As[64][H + 4];
    int tid = threadIdx.x;
    int row0 = blockIdx.x * 64;
    for (int i = tid; i < 64 * 32; i += 256) {  // 2048 float4 loads
        int r = i >> 5;
        int c4 = (i & 31) << 2;
        float4 t = *(const float4*)&A[(size_t)(row0 + r) * H + c4];
        As[r][c4 + 0] = t.x; As[r][c4 + 1] = t.y; As[r][c4 + 2] = t.z; As[r][c4 + 3] = t.w;
    }
    __syncthreads();
    int lane = tid & 31;
    int wv   = tid >> 5;
    int col0 = wv * 16;
    int lhi  = lane >> 4;
    int lm   = lane & 15;
    v8f acc[4] = {};
    #pragma unroll
    for (int kt = 0; kt < 4; ++kt) {
        v16h bf;
        #pragma unroll
        for (int v = 0; v < 8; ++v) {
            #pragma unroll
            for (int j = 0; j < 2; ++j) {
                int k = kt * 32 + ((v >> 2) << 4) + (lhi << 3) + ((v & 3) << 1) + j;
                bf[2 * v + j] = (_Float16)W[(size_t)k * H + col0 + lm];
            }
        }
        #pragma unroll
        for (int mt = 0; mt < 4; ++mt) {
            v16h af;
            #pragma unroll
            for (int v = 0; v < 8; ++v) {
                #pragma unroll
                for (int j = 0; j < 2; ++j) {
                    int k = kt * 32 + ((v >> 2) << 4) + (lhi << 3) + ((v & 3) << 1) + j;
                    af[2 * v + j] = (_Float16)As[mt * 16 + lm][k];
                }
            }
            acc[mt] = __builtin_amdgcn_wmma_f32_16x16x32_f16(false, af, false, bf, (short)0,
                                                             acc[mt], false, false);
        }
    }
    int nc = col0 + lm;
    float b = bias[nc];
    #pragma unroll
    for (int mt = 0; mt < 4; ++mt) {
        #pragma unroll
        for (int r = 0; r < 8; ++r) {
            int mr = row0 + mt * 16 + r + (lhi << 3);
            C[(size_t)mr * H + nc] = acc[mt][r] + b;
        }
    }
}

// ---------------- fused gated MLP: C = sigmoid(A@W1+b1) * (A@W2+b2) ----------------
__global__ __launch_bounds__(256) void gemm128_gated_wmma(const float* __restrict__ A,
                                                          const float* __restrict__ W1,
                                                          const float* __restrict__ b1,
                                                          const float* __restrict__ W2,
                                                          const float* __restrict__ b2,
                                                          float* __restrict__ C, int M) {
    __shared__ float As[64][H + 4];
    int tid = threadIdx.x;
    int row0 = blockIdx.x * 64;
    for (int i = tid; i < 64 * 32; i += 256) {
        int r = i >> 5;
        int c4 = (i & 31) << 2;
        float4 t = *(const float4*)&A[(size_t)(row0 + r) * H + c4];
        As[r][c4 + 0] = t.x; As[r][c4 + 1] = t.y; As[r][c4 + 2] = t.z; As[r][c4 + 3] = t.w;
    }
    __syncthreads();
    int lane = tid & 31;
    int wv   = tid >> 5;
    int col0 = wv * 16;
    int lhi  = lane >> 4;
    int lm   = lane & 15;
    v8f acc1[4] = {}, acc2[4] = {};
    #pragma unroll
    for (int kt = 0; kt < 4; ++kt) {
        v16h bf1, bf2;
        #pragma unroll
        for (int v = 0; v < 8; ++v) {
            #pragma unroll
            for (int j = 0; j < 2; ++j) {
                int k = kt * 32 + ((v >> 2) << 4) + (lhi << 3) + ((v & 3) << 1) + j;
                bf1[2 * v + j] = (_Float16)W1[(size_t)k * H + col0 + lm];
                bf2[2 * v + j] = (_Float16)W2[(size_t)k * H + col0 + lm];
            }
        }
        #pragma unroll
        for (int mt = 0; mt < 4; ++mt) {
            v16h af;
            #pragma unroll
            for (int v = 0; v < 8; ++v) {
                #pragma unroll
                for (int j = 0; j < 2; ++j) {
                    int k = kt * 32 + ((v >> 2) << 4) + (lhi << 3) + ((v & 3) << 1) + j;
                    af[2 * v + j] = (_Float16)As[mt * 16 + lm][k];
                }
            }
            acc1[mt] = __builtin_amdgcn_wmma_f32_16x16x32_f16(false, af, false, bf1, (short)0,
                                                              acc1[mt], false, false);
            acc2[mt] = __builtin_amdgcn_wmma_f32_16x16x32_f16(false, af, false, bf2, (short)0,
                                                              acc2[mt], false, false);
        }
    }
    int nc = col0 + lm;
    float bb1 = b1[nc], bb2 = b2[nc];
    #pragma unroll
    for (int mt = 0; mt < 4; ++mt) {
        #pragma unroll
        for (int r = 0; r < 8; ++r) {
            int mr = row0 + mt * 16 + r + (lhi << 3);
            float g   = sigmoidf_(acc1[mt][r] + bb1);
            float hid = acc2[mt][r] + bb2;
            C[(size_t)mr * H + nc] = g * hid;
        }
    }
}

// ---------------- per-node dots: ps = h2 . w_s, pd = h2 . w_d (wave per node) ----------------
__global__ __launch_bounds__(256) void nodedot_kernel(const float* __restrict__ h2,
                                                      const float* __restrict__ wsv,
                                                      const float* __restrict__ wdv,
                                                      float* __restrict__ ps,
                                                      float* __restrict__ pd, int n) {
    int wid = (blockIdx.x * blockDim.x + threadIdx.x) >> 5;
    if (wid >= n) return;
    int lane = threadIdx.x & 31;
    float4 hv = *(const float4*)&h2[(size_t)wid * H + lane * 4];
    float4 a = *(const float4*)&wsv[lane * 4];
    float4 b = *(const float4*)&wdv[lane * 4];
    float s = hv.x * a.x + hv.y * a.y + hv.z * a.z + hv.w * a.w;
    float d = hv.x * b.x + hv.y * b.y + hv.z * b.z + hv.w * b.w;
    #pragma unroll
    for (int m = 16; m >= 1; m >>= 1) {
        s += __shfl_xor(s, m, 32);
        d += __shfl_xor(d, m, 32);
    }
    if (lane == 0) { ps[wid] = s; pd[wid] = d; }
}

// ---------------- fused node-centric attention: softmax + aggregate + residual + leaky ----------------
// wave per node; edges bucketed by src via order/nodeoff. No atomics, fully in-register.
__global__ __launch_bounds__(256) void node_attn_kernel(const int* __restrict__ dst,
                                                        const int* __restrict__ order,
                                                        const int* __restrict__ nodeoff,
                                                        const float* __restrict__ edot,
                                                        const float* __restrict__ ps,
                                                        const float* __restrict__ pd,
                                                        const float* __restrict__ blb, int layer,
                                                        const float* __restrict__ h2,
                                                        float* __restrict__ h, int n) {
    int wid = (blockIdx.x * blockDim.x + threadIdx.x) >> 5;
    if (wid >= n) return;
    int lane = threadIdx.x & 31;
    int start = nodeoff[wid], end = nodeoff[wid + 1];
    float base = ps[wid] + blb[layer];
    // pass 1: segment max (order-independent)
    float m = -3.402823466e38f;
    for (int i = start; i < end; ++i) {
        int e = order[i];
        float x = leakyf(edot[e] + pd[dst[e]] + base);
        m = fmaxf(m, x);
    }
    // pass 2: exp-sum + weighted aggregation of h2[dst]
    float zs = 0.f;
    float4 acc = {0.f, 0.f, 0.f, 0.f};
    for (int i = start; i < end; ++i) {
        int e = order[i];
        int d = dst[e];
        float x = leakyf(edot[e] + pd[d] + base);
        float w = __expf(x - m);
        zs += w;
        float4 hv = *(const float4*)&h2[(size_t)d * H + lane * 4];
        acc.x += w * hv.x; acc.y += w * hv.y; acc.z += w * hv.z; acc.w += w * hv.w;
    }
    float inv = (end > start) ? 1.0f / zs : 0.0f;
    float4 hs = *(const float4*)&h2[(size_t)wid * H + lane * 4];
    float4 o;
    o.x = leakyf(hs.x + acc.x * inv);
    o.y = leakyf(hs.y + acc.y * inv);
    o.z = leakyf(hs.z + acc.z * inv);
    o.w = leakyf(hs.w + acc.w * inv);
    *(float4*)&h[(size_t)wid * H + lane * 4] = o;
}

// ---------------- pooling: contiguous node ranges per graph, no atomics ----------------
__global__ __launch_bounds__(128) void pool_mean_kernel(const float* __restrict__ hf,
                                                        const int* __restrict__ goffs,
                                                        float* __restrict__ graph, int G) {
    int g = blockIdx.x;
    int t = threadIdx.x;  // 0..127 = column
    int s = goffs[g], e = goffs[g + 1];
    float acc = 0.f;
    for (int node = s; node < e; ++node) acc += hf[(size_t)node * H + t];
    int cnt = e - s;
    graph[(size_t)g * H + t] = (cnt > 0) ? acc / (float)cnt : 0.f;
}

// ---------------- final: out = graph @ final_W + final_b ----------------
__global__ void final_kernel(const float* __restrict__ graph,
                             const float* __restrict__ W, const float* __restrict__ b,
                             float* __restrict__ out, int G, int C) {
    int t = blockIdx.x * blockDim.x + threadIdx.x;
    if (t >= G * C) return;
    int g = t / C, c = t % C;
    float acc = 0.f;
    for (int k = 0; k < H; ++k) acc += graph[(size_t)g * H + k] * W[k * C + c];
    out[t] = acc + b[c];
}

extern "C" void kernel_launch(void* const* d_in, const int* in_sizes, int n_in,
                              void* d_out, int out_size, void* d_ws, size_t ws_size,
                              hipStream_t stream) {
    const float* node_attr  = (const float*)d_in[0];
    const float* edge_attr  = (const float*)d_in[1];
    const int*   edge_index = (const int*)d_in[2];
    const int*   num_atoms  = (const int*)d_in[3];
    const float* enc_node_W = (const float*)d_in[4];
    const float* enc_node_b = (const float*)d_in[5];
    const float* enc_edge_W = (const float*)d_in[6];
    const float* enc_edge_b = (const float*)d_in[7];
    const float* blk_node_W = (const float*)d_in[8];
    const float* blk_node_b = (const float*)d_in[9];
    const float* blk_edge_w = (const float*)d_in[10];
    const float* blk_edge_b = (const float*)d_in[11];
    const float* gate_W1 = (const float*)d_in[12];
    const float* gate_b1 = (const float*)d_in[13];
    const float* gate_W2 = (const float*)d_in[14];
    const float* gate_b2 = (const float*)d_in[15];
    const float* gate_W3 = (const float*)d_in[16];
    const float* gate_b3 = (const float*)d_in[17];
    const float* final_W = (const float*)d_in[18];
    const float* final_b = (const float*)d_in[19];

    const int N = in_sizes[0] / H;   // 40000
    const int E = in_sizes[1] / 64;  // 512000
    const int G = in_sizes[3];       // 80
    const int L = in_sizes[11];      // 3
    const int C = in_sizes[19];      // 10

    const int* srcp = edge_index;
    const int* dstp = edge_index + E;

    // workspace carve-up (256B aligned chunks)
    char* wp = (char*)d_ws;
    auto carve = [&](size_t bytes) { void* r = (void*)wp; wp += (bytes + 255) & ~(size_t)255; return r; };
    float* h       = (float*)carve((size_t)N * H * 4);
    float* h2      = (float*)carve((size_t)N * H * 4);
    float* tbuf    = (float*)carve((size_t)N * H * 4);
    float* edot    = (float*)carve((size_t)E * 3 * 4);  // [l][E]
    float* ps      = (float*)carve((size_t)N * 4);
    float* pd      = (float*)carve((size_t)N * 4);
    int*   deg     = (int*)carve((size_t)N * 4);
    int*   nodeoff = (int*)carve((size_t)(N + 1) * 4);
    int*   cursor  = (int*)carve((size_t)N * 4);
    int*   order   = (int*)carve((size_t)E * 4);
    int*   bsum    = (int*)carve(4096);
    float* Q       = (float*)carve(3 * 64 * 4);
    float* cvec    = (float*)carve(16);
    int*   goffs   = (int*)carve((size_t)(G + 1) * 4);
    float* graph   = (float*)carve((size_t)G * H * 4);

    const int TB = 256;
    auto blks = [](long long n, int tb) { return (int)((n + tb - 1) / tb); };
    const int nScanBlocks = blks(N, 256);

    // ---- once-only prep ----
    prep_kernel<<<1, 256, 0, stream>>>(enc_edge_W, enc_edge_b, blk_edge_w, num_atoms,
                                       Q, cvec, goffs, G, L);
    edot_kernel<<<blks((long long)E * 16, TB), TB, 0, stream>>>(edge_attr, Q, cvec, edot, E);

    // bucket edges by src (reused across all layers)
    fill_u32_kernel<<<blks(N, TB), TB, 0, stream>>>((unsigned*)deg, 0u, N);
    count_kernel<<<blks(E, TB), TB, 0, stream>>>(srcp, deg, E);
    scanA_kernel<<<nScanBlocks, 256, 0, stream>>>(deg, nodeoff, bsum, N);
    scanB_kernel<<<1, 32, 0, stream>>>(bsum, nScanBlocks);
    scanC_kernel<<<nScanBlocks, 256, 0, stream>>>(nodeoff, bsum, N, E);
    copy_i32_kernel<<<blks(N, TB), TB, 0, stream>>>(nodeoff, cursor, N);
    scatter_kernel<<<blks(E, TB), TB, 0, stream>>>(srcp, cursor, order, E);

    // encoder: h = node_attr @ enc_node_W + b
    gemm128_wmma<<<N / 64, TB, 0, stream>>>(node_attr, enc_node_W, enc_node_b, h, N);

    for (int l = 0; l < L; ++l) {
        gemm128_wmma<<<N / 64, TB, 0, stream>>>(h, blk_node_W + (size_t)l * H * H,
                                                blk_node_b + (size_t)l * H, h2, N);
        nodedot_kernel<<<blks((long long)N * 32, TB), TB, 0, stream>>>(
            h2, blk_edge_w + (size_t)l * 3 * H + H, blk_edge_w + (size_t)l * 3 * H + 2 * H,
            ps, pd, N);
        node_attn_kernel<<<blks((long long)N * 32, TB), TB, 0, stream>>>(
            dstp, order, nodeoff, edot + (size_t)l * E, ps, pd, blk_edge_b, l, h2, h, N);
    }

    // gated MLP: t = sigmoid(h@W1+b1) * (h@W2+b2) -> tbuf ; h2 = t@W3+b3
    gemm128_gated_wmma<<<N / 64, TB, 0, stream>>>(h, gate_W1, gate_b1, gate_W2, gate_b2, tbuf, N);
    gemm128_wmma<<<N / 64, TB, 0, stream>>>(tbuf, gate_W3, gate_b3, h2, N);

    // pooling (contiguous segments) + final projection
    pool_mean_kernel<<<G, 128, 0, stream>>>(h2, goffs, graph, G);
    final_kernel<<<blks((long long)G * C, TB), TB, 0, stream>>>(graph, final_W, final_b,
                                                                (float*)d_out, G, C);
}
